// SwinTransformerBlock_15023795601920
// MI455X (gfx1250) — compile-verified, matrix operand order checked by reading, not run on hardware
//
#include <hip/hip_runtime.h>
#include <hip/hip_bf16.h>

typedef _Float16 half_t;
typedef __attribute__((ext_vector_type(16))) _Float16 v16h;
typedef __attribute__((ext_vector_type(8)))  _Float16 v8h;
typedef __attribute__((ext_vector_type(8)))  float    v8f;
typedef __attribute__((ext_vector_type(4)))  int      v4i;

#define DIMC 192
#define HEADS 6
#define HEAD_DIM 32
#define WSZ 8
#define NTOK 64
#define BIMG 16
#define HIMG 56
#define NWIN 49                 // 7*7 windows per image
#define BN   (BIMG * NWIN)      // 784 windows total
#define MROWS (BN * NTOK)       // 50176 tokens
#define MLP_HID 768
#define PLD 72                  // padded LDS leading dim for attention (144B rows, 16B aligned)

#if __has_builtin(__builtin_amdgcn_global_load_async_to_lds_b128) && __has_builtin(__builtin_amdgcn_s_wait_asynccnt)
#define HAS_ASYNC_LDS 1
typedef __attribute__((address_space(1))) v4i* gas_v4i_ptr;
typedef __attribute__((address_space(3))) v4i* las_v4i_ptr;
#else
#define HAS_ASYNC_LDS 0
#endif

__device__ __forceinline__ v16h load_frag16(const half_t* p) {
  // A/B fragment (16-bit, 16x32 / 32x16): halves 0..7 = K k0..k0+7, halves 8..15 = K k0+16..k0+23
  v8h lo = *(const v8h*)(p);
  v8h hi = *(const v8h*)(p + 16);
  return __builtin_shufflevector(lo, hi, 0,1,2,3,4,5,6,7,8,9,10,11,12,13,14,15);
}

__device__ __forceinline__ v8f zero8() {
  v8f z = {0.f,0.f,0.f,0.f,0.f,0.f,0.f,0.f};
  return z;
}

__device__ __forceinline__ int region3(int c) {
  // Swin shifted-window region id along one axis: slices (0,-8), (-8,-4), (-4,None)
  return (c < HIMG - WSZ) ? 0 : ((c < HIMG - 4) ? 1 : 2);
}

// Stage the block's 64-row weight tile Wt[col0..col0+63][0..K) into padded LDS
// (row stride K+8 halves for bank spread). Uses CDNA5 async global->LDS when available.
template <int K>
__device__ __forceinline__ void stage_b_tile(const half_t* __restrict__ Wt, int col0,
                                             half_t* Bs) {
  const int chunks_per_row = K / 8;            // 16B chunks
  const int nchunks = 64 * chunks_per_row;
  for (int c = threadIdx.x; c < nchunks; c += blockDim.x) {
    const int row = c / chunks_per_row;
    const int cc  = c - row * chunks_per_row;
    const half_t* g = Wt + (size_t)(col0 + row) * K + cc * 8;
    half_t*       l = Bs + row * (K + 8) + cc * 8;
#if HAS_ASYNC_LDS
    __builtin_amdgcn_global_load_async_to_lds_b128((gas_v4i_ptr)g, (las_v4i_ptr)l, 0, 0);
#else
    *(v8h*)l = *(const v8h*)g;
#endif
  }
#if HAS_ASYNC_LDS
  __builtin_amdgcn_s_wait_asynccnt(0);
#endif
  __syncthreads();
}

// Main loop: acc[4] += A(16xK from global) x B(Kx64 from LDS tile)
template <int K>
__device__ __forceinline__ void wmma_mainloop(const half_t* __restrict__ arow,
                                              const half_t* __restrict__ Bs,
                                              int mlane, int k0, v8f acc[4]) {
  const half_t* brow = Bs + mlane * (K + 8) + k0;
#pragma unroll
  for (int kt = 0; kt < K; kt += 32) {
    v16h af = load_frag16(arow + kt);
#pragma unroll
    for (int t = 0; t < 4; ++t) {
      v16h bf = load_frag16(brow + (16 * t) * (K + 8) + kt);
      acc[t] = __builtin_amdgcn_wmma_f32_16x16x32_f16(false, af, false, bf, (short)0, acc[t], false, false);
    }
  }
}

// ---------------- weight convert: W[K][N] f32 -> Wt[N][K] f16 ----------------
__global__ void convert_wt(const float* __restrict__ W, half_t* __restrict__ Wt, int K, int N) {
  int idx = blockIdx.x * 256 + threadIdx.x;
  if (idx < K * N) {
    int n = idx / K, k = idx - n * K;
    Wt[idx] = (half_t)W[k * N + n];
  }
}

// ---------------- LN1 + roll(-4,-4) + window partition -> f16 ----------------
__global__ void ln1_shift_partition(const float* __restrict__ x,
                                    const float* __restrict__ gamma,
                                    const float* __restrict__ beta,
                                    half_t* __restrict__ xw) {
  const int lane = threadIdx.x & 31;
  const int wave = threadIdx.x >> 5;
  const int tk = blockIdx.x * 8 + wave;            // window-token id 0..50175
  const int bn = tk >> 6, t = tk & 63;
  const int b  = bn / NWIN, w = bn % NWIN;
  const int wy = w / 7, wx = w % 7;
  int hs = wy * WSZ + (t >> 3) + 4; if (hs >= HIMG) hs -= HIMG;  // roll(-4): src = (i+4)%56
  int wsx = wx * WSZ + (t & 7) + 4; if (wsx >= HIMG) wsx -= HIMG;
  const float* row = x + ((size_t)(b * HIMG + hs) * HIMG + wsx) * DIMC;
  float v[6], s = 0.f, sq = 0.f;
#pragma unroll
  for (int i = 0; i < 6; ++i) { v[i] = row[lane + 32 * i]; s += v[i]; sq += v[i] * v[i]; }
#pragma unroll
  for (int off = 1; off < 32; off <<= 1) { s += __shfl_xor(s, off, 32); sq += __shfl_xor(sq, off, 32); }
  float mean = s * (1.f / DIMC);
  float rstd = rsqrtf(sq * (1.f / DIMC) - mean * mean + 1e-5f);
  half_t* orow = xw + (size_t)tk * DIMC;
#pragma unroll
  for (int i = 0; i < 6; ++i) {
    int c = lane + 32 * i;
    orow[c] = (half_t)((v[i] - mean) * rstd * gamma[c] + beta[c]);
  }
}

// ---------------- LN2 over contiguous rows -> f16 ----------------
__global__ void ln2_rows(const float* __restrict__ x1,
                         const float* __restrict__ gamma,
                         const float* __restrict__ beta,
                         half_t* __restrict__ xn) {
  const int lane = threadIdx.x & 31;
  const int wave = threadIdx.x >> 5;
  const int tk = blockIdx.x * 8 + wave;
  const float* row = x1 + (size_t)tk * DIMC;
  float v[6], s = 0.f, sq = 0.f;
#pragma unroll
  for (int i = 0; i < 6; ++i) { v[i] = row[lane + 32 * i]; s += v[i]; sq += v[i] * v[i]; }
#pragma unroll
  for (int off = 1; off < 32; off <<= 1) { s += __shfl_xor(s, off, 32); sq += __shfl_xor(sq, off, 32); }
  float mean = s * (1.f / DIMC);
  float rstd = rsqrtf(sq * (1.f / DIMC) - mean * mean + 1e-5f);
  half_t* orow = xn + (size_t)tk * DIMC;
#pragma unroll
  for (int i = 0; i < 6; ++i) {
    int c = lane + 32 * i;
    orow[c] = (half_t)((v[i] - mean) * rstd * gamma[c] + beta[c]);
  }
}

// ---------------- generic WMMA GEMM: C f16 = act(A[M][K]f16 @ Wt[N][K]^T + bias) ----------------
// block: 256 thr = 8 waves; wave -> 16 rows x 64 cols; grid (N/64, M/128); B tile staged in LDS
template <bool GELU, int K, int N>
__global__ void gemm_f16(const half_t* __restrict__ A, const half_t* __restrict__ Wt,
                         const float* __restrict__ bias, half_t* __restrict__ C) {
  __shared__ half_t Bs[64 * (K + 8)];
  const int lane  = threadIdx.x & 31;
  const int wave  = threadIdx.x >> 5;
  const int mlane = lane & 15;
  const int k0    = (lane >> 4) ? 8 : 0;
  const int row0  = blockIdx.y * 128 + wave * 16;
  const int col0  = blockIdx.x * 64;
  stage_b_tile<K>(Wt, col0, Bs);
  v8f acc[4];
#pragma unroll
  for (int t = 0; t < 4; ++t) acc[t] = zero8();
  const half_t* arow = A + (size_t)(row0 + mlane) * K + k0;
  wmma_mainloop<K>(arow, Bs, mlane, k0, acc);
  const int rbase = row0 + ((lane >> 4) << 3);
#pragma unroll
  for (int t = 0; t < 4; ++t) {
    const int col = col0 + 16 * t + mlane;
    const float bv = bias[col];
#pragma unroll
    for (int r = 0; r < 8; ++r) {
      float val = acc[t][r] + bv;
      if (GELU) val = 0.5f * val * (1.f + erff(val * 0.70710678118654752f));
      C[(size_t)(rbase + r) * N + col] = (half_t)val;
    }
  }
}

// ---------------- proj GEMM + window merge + roll(+4,+4) + residual -> x1 f32 ----------------
__global__ void gemm_proj(const half_t* __restrict__ A, const half_t* __restrict__ Wt,
                          const float* __restrict__ bias, const float* __restrict__ xin,
                          float* __restrict__ x1) {
  const int K = DIMC;
  __shared__ half_t Bs[64 * (DIMC + 8)];
  const int lane  = threadIdx.x & 31;
  const int wave  = threadIdx.x >> 5;
  const int mlane = lane & 15;
  const int k0    = (lane >> 4) ? 8 : 0;
  const int row0  = blockIdx.y * 128 + wave * 16;
  const int col0  = blockIdx.x * 64;
  stage_b_tile<DIMC>(Wt, col0, Bs);
  v8f acc[4];
#pragma unroll
  for (int t = 0; t < 4; ++t) acc[t] = zero8();
  const half_t* arow = A + (size_t)(row0 + mlane) * K + k0;
  wmma_mainloop<DIMC>(arow, Bs, mlane, k0, acc);
  const int rbase = row0 + ((lane >> 4) << 3);
#pragma unroll
  for (int t = 0; t < 4; ++t) {
    const int col = col0 + 16 * t + mlane;
    const float bv = bias[col];
#pragma unroll
    for (int r = 0; r < 8; ++r) {
      const int m  = rbase + r;
      const int bn = m >> 6, tok = m & 63;
      const int b  = bn / NWIN, w = bn % NWIN;
      const int wy = w / 7, wx = w % 7;
      int hs  = wy * WSZ + (tok >> 3) + 4; if (hs  >= HIMG) hs  -= HIMG;  // roll back
      int wsx = wx * WSZ + (tok & 7)  + 4; if (wsx >= HIMG) wsx -= HIMG;
      const size_t orow = ((size_t)(b * HIMG + hs) * HIMG + wsx) * DIMC;
      x1[orow + col] = xin[orow + col] + acc[t][r] + bv;
    }
  }
}

// ---------------- MLP2 GEMM + bias + residual(x1) -> f32 out ----------------
__global__ void gemm_mlp2(const half_t* __restrict__ A, const half_t* __restrict__ Wt,
                          const float* __restrict__ bias, const float* __restrict__ x1,
                          float* __restrict__ out) {
  const int K = MLP_HID, N = DIMC;
  __shared__ half_t Bs[64 * (MLP_HID + 8)];
  const int lane  = threadIdx.x & 31;
  const int wave  = threadIdx.x >> 5;
  const int mlane = lane & 15;
  const int k0    = (lane >> 4) ? 8 : 0;
  const int row0  = blockIdx.y * 128 + wave * 16;
  const int col0  = blockIdx.x * 64;
  stage_b_tile<MLP_HID>(Wt, col0, Bs);
  v8f acc[4];
#pragma unroll
  for (int t = 0; t < 4; ++t) acc[t] = zero8();
  const half_t* arow = A + (size_t)(row0 + mlane) * K + k0;
  wmma_mainloop<MLP_HID>(arow, Bs, mlane, k0, acc);
  const int rbase = row0 + ((lane >> 4) << 3);
#pragma unroll
  for (int t = 0; t < 4; ++t) {
    const int col = col0 + 16 * t + mlane;
    const float bv = bias[col];
#pragma unroll
    for (int r = 0; r < 8; ++r) {
      const size_t m = (size_t)(rbase + r);
      out[m * N + col] = x1[m * N + col] + acc[t][r] + bv;
    }
  }
}

// ---------------- windowed attention: one block per (window, head), 4 waves ----------------
__global__ void window_attention(const half_t* __restrict__ qkv,
                                 const float* __restrict__ rel_table,
                                 half_t* __restrict__ attn_out) {
  __shared__ half_t probs[NTOK * PLD];     // softmax(QK^T) in f16
  __shared__ half_t vT[HEAD_DIM * PLD];    // v transposed: vT[d][token]
  const int bn    = blockIdx.x;
  const int h     = blockIdx.y;
  const int lane  = threadIdx.x & 31;
  const int wave  = threadIdx.x >> 5;      // 0..3
  const int mlane = lane & 15;
  const int hhi   = lane >> 4;             // 0 or 1 (lane half)
  const int k0    = hhi ? 8 : 0;
  const int m0    = wave * 16;
  const half_t* qk = qkv + (size_t)bn * NTOK * 576;

  // stage v^T into LDS (v = qkv channels [384 + h*32 .. +32))
  for (int idx = threadIdx.x; idx < NTOK * HEAD_DIM; idx += blockDim.x) {
    int tok = idx >> 5, d = idx & 31;
    vT[d * PLD + tok] = qk[tok * 576 + 384 + h * 32 + d];
  }

  // scores = q @ k^T : wave computes 16 rows x 64 cols, K=32 (one WMMA step per tile)
  v8f acc[4];
#pragma unroll
  for (int t = 0; t < 4; ++t) acc[t] = zero8();
  v16h qf = load_frag16(qk + (m0 + mlane) * 576 + h * 32 + k0);
#pragma unroll
  for (int t = 0; t < 4; ++t) {
    v16h kf = load_frag16(qk + (16 * t + mlane) * 576 + 192 + h * 32 + k0);
    acc[t] = __builtin_amdgcn_wmma_f32_16x16x32_f16(false, qf, false, kf, (short)0, acc[t], false, false);
  }

  // scale + relative-position bias + shifted-window mask, then row softmax
  const int wgi = bn % NWIN;
  const int wy = wgi / 7, wx = wgi % 7;
  int yj[4], xj[4], rj[4];
#pragma unroll
  for (int t = 0; t < 4; ++t) {
    const int j = 16 * t + mlane;
    yj[t] = j >> 3; xj[t] = j & 7;
    rj[t] = 3 * region3(wy * WSZ + yj[t]) + region3(wx * WSZ + xj[t]);
  }
  const float scale = 0.1767766952966369f;  // 1/sqrt(32)
  float s[4][8];
#pragma unroll
  for (int r = 0; r < 8; ++r) {
    const int m = m0 + r + 8 * hhi;
    const int yi = m >> 3, xi = m & 7;
    const int ri = 3 * region3(wy * WSZ + yi) + region3(wx * WSZ + xi);
#pragma unroll
    for (int t = 0; t < 4; ++t) {
      const int rel = (yi - yj[t] + 7) * 15 + (xi - xj[t] + 7);
      const float bv = rel_table[rel * HEADS + h];
      s[t][r] = acc[t][r] * scale + bv + ((ri == rj[t]) ? 0.f : -100.f);
    }
  }
#pragma unroll
  for (int r = 0; r < 8; ++r) {
    float mx = s[0][r];
#pragma unroll
    for (int t = 1; t < 4; ++t) mx = fmaxf(mx, s[t][r]);
#pragma unroll
    for (int off = 1; off < 16; off <<= 1) mx = fmaxf(mx, __shfl_xor(mx, off, 32));
    float sum = 0.f;
#pragma unroll
    for (int t = 0; t < 4; ++t) { s[t][r] = __expf(s[t][r] - mx); sum += s[t][r]; }
#pragma unroll
    for (int off = 1; off < 16; off <<= 1) sum += __shfl_xor(sum, off, 32);
    const float inv = 1.f / sum;
    const int m = m0 + r + 8 * hhi;
#pragma unroll
    for (int t = 0; t < 4; ++t) probs[m * PLD + 16 * t + mlane] = (half_t)(s[t][r] * inv);
  }
  __syncthreads();

  // out = probs(64x64) @ v(64x32): 2 K-steps x 2 N-tiles per wave
  v8f o[2];
  o[0] = zero8(); o[1] = zero8();
#pragma unroll
  for (int kt = 0; kt < 64; kt += 32) {
    v16h pf = load_frag16(&probs[(m0 + mlane) * PLD + kt + k0]);
#pragma unroll
    for (int t = 0; t < 2; ++t) {
      v16h vf = load_frag16(&vT[(16 * t + mlane) * PLD + kt + k0]);
      o[t] = __builtin_amdgcn_wmma_f32_16x16x32_f16(false, pf, false, vf, (short)0, o[t], false, false);
    }
  }
#pragma unroll
  for (int t = 0; t < 2; ++t) {
#pragma unroll
    for (int r = 0; r < 8; ++r) {
      const int m = m0 + r + 8 * hhi;
      attn_out[((size_t)bn * NTOK + m) * DIMC + h * 32 + 16 * t + mlane] = (half_t)o[t][r];
    }
  }
}

extern "C" void kernel_launch(void* const* d_in, const int* in_sizes, int n_in,
                              void* d_out, int out_size, void* d_ws, size_t ws_size,
                              hipStream_t stream) {
  (void)in_sizes; (void)n_in; (void)out_size; (void)ws_size;
  const float* x       = (const float*)d_in[0];
  const float* n1g     = (const float*)d_in[1];
  const float* n1b     = (const float*)d_in[2];
  const float* qkv_w   = (const float*)d_in[3];
  const float* qkv_b   = (const float*)d_in[4];
  const float* proj_w  = (const float*)d_in[5];
  const float* proj_b  = (const float*)d_in[6];
  const float* rel_tab = (const float*)d_in[7];
  const float* n2g     = (const float*)d_in[8];
  const float* n2b     = (const float*)d_in[9];
  const float* w1      = (const float*)d_in[10];
  const float* b1      = (const float*)d_in[11];
  const float* w2      = (const float*)d_in[12];
  const float* b2      = (const float*)d_in[13];
  float* out = (float*)d_out;

  char* ws = (char*)d_ws;
  size_t off = 0;
  auto alloc = [&](size_t bytes) { size_t o = off; off += (bytes + 255) & ~(size_t)255; return o; };
  half_t* qkv_wt  = (half_t*)(ws + alloc((size_t)576 * 192 * 2));
  half_t* proj_wt = (half_t*)(ws + alloc((size_t)192 * 192 * 2));
  half_t* w1t     = (half_t*)(ws + alloc((size_t)768 * 192 * 2));
  half_t* w2t     = (half_t*)(ws + alloc((size_t)192 * 768 * 2));
  half_t* xw      = (half_t*)(ws + alloc((size_t)MROWS * DIMC * 2));     // LN1 windows (f16)
  half_t* qkvbuf  = (half_t*)(ws + alloc((size_t)MROWS * 576 * 2));      // QKV (f16)
  float*  x1      = (float*) (ws + alloc((size_t)MROWS * DIMC * 4));     // attn residual (f32)
  half_t* hidden  = (half_t*)(ws + alloc((size_t)MROWS * MLP_HID * 2));  // MLP hidden (f16)
  half_t* attn_o  = xw;      // alias: xw dead after QKV GEMM
  half_t* xn2     = qkvbuf;  // alias: qkv dead after attention

  // 1) convert+transpose weights to f16 [N][K]
  convert_wt<<<dim3((192 * 576 + 255) / 256), 256, 0, stream>>>(qkv_w, qkv_wt, 192, 576);
  convert_wt<<<dim3((192 * 192 + 255) / 256), 256, 0, stream>>>(proj_w, proj_wt, 192, 192);
  convert_wt<<<dim3((192 * 768 + 255) / 256), 256, 0, stream>>>(w1, w1t, 192, 768);
  convert_wt<<<dim3((768 * 192 + 255) / 256), 256, 0, stream>>>(w2, w2t, 768, 192);

  // 2) LN1 + shift + window partition
  ln1_shift_partition<<<dim3(MROWS / 8), 256, 0, stream>>>(x, n1g, n1b, xw);

  // 3) QKV GEMM (50176x192 @ 192x576)
  gemm_f16<false, 192, 576><<<dim3(576 / 64, MROWS / 128), 256, 0, stream>>>(xw, qkv_wt, qkv_b, qkvbuf);

  // 4) windowed attention (784 windows x 6 heads)
  window_attention<<<dim3(BN, HEADS), 128, 0, stream>>>(qkvbuf, rel_tab, attn_o);

  // 5) proj GEMM + merge + roll-back + residual -> x1
  gemm_proj<<<dim3(192 / 64, MROWS / 128), 256, 0, stream>>>(attn_o, proj_wt, proj_b, x, x1);

  // 6) LN2
  ln2_rows<<<dim3(MROWS / 8), 256, 0, stream>>>(x1, n2g, n2b, xn2);

  // 7) MLP1 + GELU
  gemm_f16<true, 192, 768><<<dim3(768 / 64, MROWS / 128), 256, 0, stream>>>(xn2, w1t, b1, hidden);

  // 8) MLP2 + residual -> out
  gemm_mlp2<<<dim3(192 / 64, MROWS / 128), 256, 0, stream>>>(hidden, w2t, b2, x1, out);
}
